// SelfMultiheadAttention_84679575208543
// MI455X (gfx1250) — matrix operand order repaired and checked by reading, not compiled, not run on hardware
//
#include <hip/hip_runtime.h>

typedef __bf16 bf16;
typedef __attribute__((ext_vector_type(16))) __bf16 v16bf;
typedef __attribute__((ext_vector_type(4)))  __bf16 v4bf;
typedef __attribute__((ext_vector_type(8)))  float  v8f;
typedef unsigned int u32;
typedef __attribute__((ext_vector_type(4))) u32 v4u;
typedef __attribute__((ext_vector_type(8))) int v8i;
typedef __attribute__((ext_vector_type(4))) int v4i;

#define E_DIM   1024
#define N_NODE  512
#define N_GRAPH 8
#define NHEADS  16
#define HD      64
#define NBATCH  128      // n_graph * H
#define M_ROWS  4096     // n_node * n_graph
#define QKV_LD  3072
#define SCALING 0.125f   // 64^-0.5

// ---------------------------------------------------------------------------
// Fragment loaders, contiguous k = 16*hi + i in A and B (wave32, 16-bit A 16x32 / B 32x16)
// ---------------------------------------------------------------------------

// A[M,K] row-major with row stride ld.  K(i) = k0 + 16*hi + i
__device__ __forceinline__ v16bf load_a_rowmajor(const bf16* __restrict__ base,
                                                 int row0, int ld, int k0) {
  const int lane = threadIdx.x & 31;
  const int m  = row0 + (lane & 15);
  const int hi = lane >> 4;
  const bf16* p = base + (long)m * ld + k0 + hi * 16;
  v16bf a;
#pragma unroll
  for (int g = 0; g < 2; ++g)
#pragma unroll
    for (int j = 0; j < 8; ++j)
      a[g * 8 + j] = p[g * 8 + j];             // one 32B contiguous run
  return a;
}

// B(K,N) = mat[N*ld + K]   (B supplied as row-major [N x K], e.g. W of x@W.T)
__device__ __forceinline__ v16bf load_b_nk(const bf16* __restrict__ base,
                                           int col0, int ld, int k0) {
  const int lane = threadIdx.x & 31;
  const int n  = col0 + (lane & 15);
  const int hi = lane >> 4;
  const bf16* p = base + (long)n * ld + k0 + hi * 16;
  v16bf b;
#pragma unroll
  for (int i = 0; i < 16; ++i) b[i] = p[i];    // K = k0 + 16*hi + i
  return b;
}

// B(K,N) from an LDS-resident [32 x 64] bf16 tile (row-major, K-major)
__device__ __forceinline__ v16bf load_b_lds_kn(const bf16* base, int col0) {
  const int lane = threadIdx.x & 31;
  const int n  = col0 + (lane & 15);
  const int hi = lane >> 4;
  v16bf b;
#pragma unroll
  for (int i = 0; i < 16; ++i) b[i] = base[(16 * hi + i) * HD + n];
  return b;
}

__device__ __forceinline__ v8f wmma_bf16(v16bf a, v16bf b, v8f c) {
  return __builtin_amdgcn_wmma_f32_16x16x32_bf16(
      /*neg_a=*/false, a, /*neg_b=*/false, b,
      /*c_mod=*/(short)0, c, /*reuse_a=*/false, /*reuse_b=*/false);
}

// ---------------------------------------------------------------------------
// TDM: async 2D tile (bf16) global -> LDS, descriptor per ISA §8.3/8.4.
// All inputs must be wave-uniform (descriptor lives in SGPRs).
// Toolchain exposes the 6-arg builtin: (v4u g0, v8i g1, v4i, v4i, v8i, cpol).
// ---------------------------------------------------------------------------
__device__ __forceinline__ void tdm_load_2d_bf16(u32 lds_off_bytes,
                                                 const void* gaddr,
                                                 u32 tile_d0, u32 tile_d1,
                                                 u32 tensor_d0, u32 tensor_d1,
                                                 u32 stride_d0_elems) {
  const unsigned long long ga = (unsigned long long)gaddr;
  v4u g0;
  g0[0] = 1u;                                   // count=1, load, user mode
  g0[1] = lds_off_bytes;                        // lds_addr
  g0[2] = (u32)(ga & 0xffffffffu);              // global_addr[31:0]
  g0[3] = (u32)((ga >> 32) & 0x01ffffffu) | (2u << 30);  // addr[56:32]|type=2
  v8i g1;
  g1[0] = (int)(1u << 16);                      // data_size=1 (2 bytes)
  g1[1] = (int)((tensor_d0 & 0xffffu) << 16);   // tensor_dim0[15:0] @ bits63:48
  g1[2] = (int)(((tensor_d0 >> 16) & 0xffffu) | ((tensor_d1 & 0xffffu) << 16));
  g1[3] = (int)(((tensor_d1 >> 16) & 0xffffu) | ((tile_d0 & 0xffffu) << 16));
  g1[4] = (int)(tile_d1 & 0xffffu);             // tile_dim1 (tile_dim2 = 0)
  g1[5] = (int)stride_d0_elems;                 // tensor_dim0_stride[31:0]
  g1[6] = 0;
  g1[7] = 0;
  v4i g2 = {0, 0, 0, 0};                        // 2D: groups 2/3 unused
  v4i g3 = {0, 0, 0, 0};
  v8i g4 = {0, 0, 0, 0, 0, 0, 0, 0};
  __builtin_amdgcn_tensor_load_to_lds(g0, g1, g2, g3, g4, 0);
}

// ---------------------------------------------------------------------------
// fp32 -> bf16 conversion (float4 -> 4x bf16 per thread)
// ---------------------------------------------------------------------------
__global__ void __launch_bounds__(256) cvt_bf16_kernel(
    const float4* __restrict__ src, v4bf* __restrict__ dst, int n4) {
  int i = blockIdx.x * 256 + threadIdx.x;
  if (i < n4) {
    float4 f = src[i];
    v4bf o;
    o[0] = (bf16)f.x; o[1] = (bf16)f.y; o[2] = (bf16)f.z; o[3] = (bf16)f.w;
    dst[i] = o;
  }
}

// ---------------------------------------------------------------------------
// QKV projection:  qkv[m, n3] = sum_k query[m,k] * in_w[n3,k] + in_b[n3]
// One wave computes a 16x64 tile: A fragment reused across 4 WMMAs per k-step.
// q third pre-scaled by hd^-0.5.  Output bf16 [4096 x 3072].
// ---------------------------------------------------------------------------
__global__ void __launch_bounds__(256) qkv_gemm_kernel(
    const bf16* __restrict__ A, const bf16* __restrict__ W,
    const float* __restrict__ bias, bf16* __restrict__ qkv) {
  const int wid = threadIdx.x >> 5;
  const int n0 = (blockIdx.x * 8 + wid) * 64;   // 0..3008
  const int m0 = blockIdx.y * 16;               // 0..4080
  v8f c[4] = {};
  for (int k0 = 0; k0 < E_DIM; k0 += 32) {
    v16bf a = load_a_rowmajor(A, m0, E_DIM, k0);
#pragma unroll
    for (int t = 0; t < 4; ++t)
      c[t] = wmma_bf16(a, load_b_nk(W, n0 + t * 16, E_DIM, k0), c[t]);
  }
  const int lane = threadIdx.x & 31;
  const int hi = lane >> 4;
#pragma unroll
  for (int t = 0; t < 4; ++t) {
    const int n = n0 + t * 16 + (lane & 15);
    const float bv  = bias[n];
    const float scl = (n < E_DIM) ? SCALING : 1.0f;
#pragma unroll
    for (int j = 0; j < 8; ++j) {
      int m = m0 + j + 8 * hi;
      qkv[(long)m * QKV_LD + n] = (bf16)((c[t][j] + bv) * scl);
    }
  }
}

// ---------------------------------------------------------------------------
// Fused attention, one wave per workgroup (descriptors stay scalar; barriers
// are S_NOP).  Online softmax; V tiles arrive via double-buffered TDM DMA.
// attn_probs = tanh(gw) * softmax(S): tanh weights the numerator only.
// LDS map: [0,1KB) P tile 16x32 bf16; [1KB,5KB) V buf0; [5KB,9KB) V buf1.
// ---------------------------------------------------------------------------
#define LDS_SP 0u
#define LDS_SV0 1024u
#define LDS_SV1 5120u

__global__ void __launch_bounds__(32) attention_kernel(
    const bf16* __restrict__ qkv, const float* __restrict__ bias,
    const float* __restrict__ gw, bf16* __restrict__ attnb) {
  __shared__ __align__(16) unsigned char lds_raw[1024 + 2 * 4096];
  bf16* const sP = (bf16*)(lds_raw + LDS_SP);       // [16][32]

  const int lane = threadIdx.x & 31;
  const int hi   = lane >> 4;
  const int lo   = lane & 15;

  const int b  = blockIdx.x >> 5;                   // batch 0..127
  const int n0 = (blockIdx.x & 31) * 16;            // query-row tile
  const int g  = b >> 4;
  const int h  = b & 15;

  const bf16* qbase = qkv + (long)g * QKV_LD + h * HD;     // q region
  const bf16* kbase = qbase + E_DIM;                       // k region
  const bf16* vbase = qbase + 2 * E_DIM;                   // v region
  const int   bld   = N_GRAPH * QKV_LD;                    // 24576

  // q fragments reused by every column step: hoist.
  const v16bf aq0 = load_a_rowmajor(qbase, n0, bld, 0);
  const v16bf aq1 = load_a_rowmajor(qbase, n0, bld, 32);

  v8f   oacc[4] = {};                               // [16 x 64] f32 accum
  float m_run[8], l_run[8];
#pragma unroll
  for (int j = 0; j < 8; ++j) { m_run[j] = -3.0e38f; l_run[j] = 0.0f; }

  // Prime the TDM pipeline: V rows 0..31 -> buf0.
  tdm_load_2d_bf16(LDS_SV0, vbase, HD, 32, HD, N_NODE, (u32)bld);

  for (int cs = 0; cs < N_NODE / 32; ++cs) {        // 16 super-steps, 32 cols
    if (cs < 15) {
      // next V tile into the other buffer (overlaps with S/softmax below)
      tdm_load_2d_bf16(((cs + 1) & 1) ? LDS_SV1 : LDS_SV0,
                       vbase + (long)(cs + 1) * 32 * bld,
                       HD, 32, HD, N_NODE, (u32)bld);
      // prefetch next bias/gw tile (32 cachelines each, one line per lane)
      const long poff = ((long)b * N_NODE + n0 + (lane >> 1)) * N_NODE +
                        (long)(cs + 1) * 32 + (lane & 1) * 16;
      __builtin_prefetch(&bias[poff], 0, 0);
      __builtin_prefetch(&gw[poff], 0, 0);
    }

    float st[2][8], gt[2][8];
#pragma unroll
    for (int t = 0; t < 2; ++t) {
      const int col0 = cs * 32 + t * 16;
      v8f c = {};
      c = wmma_bf16(aq0, load_b_nk(kbase, col0, bld, 0),  c);
      c = wmma_bf16(aq1, load_b_nk(kbase, col0, bld, 32), c);
#pragma unroll
      for (int j = 0; j < 8; ++j) {
        const int n  = n0 + j + 8 * hi;
        const int mc = col0 + lo;
        const long off = ((long)b * N_NODE + n) * N_NODE + mc;
        st[t][j] = c[j] + bias[off];
        gt[t][j] = tanhf(gw[off]);
      }
    }

    // online softmax update (rows live in 16-lane halves)
#pragma unroll
    for (int j = 0; j < 8; ++j) {
      float rmax = fmaxf(st[0][j], st[1][j]);
#pragma unroll
      for (int msk = 1; msk < 16; msk <<= 1)
        rmax = fmaxf(rmax, __shfl_xor(rmax, msk, 32));
      const float mnew = fmaxf(m_run[j], rmax);
      const float resc = __expf(m_run[j] - mnew);
      m_run[j] = mnew;
      l_run[j] *= resc;
      oacc[0][j] *= resc; oacc[1][j] *= resc;
      oacc[2][j] *= resc; oacc[3][j] *= resc;

      const float p0 = __expf(st[0][j] - mnew);
      const float p1 = __expf(st[1][j] - mnew);
      float ls = p0 + p1;
#pragma unroll
      for (int msk = 1; msk < 16; msk <<= 1)
        ls += __shfl_xor(ls, msk, 32);
      l_run[j] += ls;

      sP[(j + 8 * hi) * 32 + lo]      = (bf16)(gt[0][j] * p0);
      sP[(j + 8 * hi) * 32 + 16 + lo] = (bf16)(gt[1][j] * p1);
    }
    __syncthreads();   // single-wave WG: S_NOP in HW, compiler memory fence

    // P tile (C-layout in LDS) -> A-layout fragment
    v16bf ap;
#pragma unroll
    for (int gg = 0; gg < 2; ++gg)
#pragma unroll
      for (int jj = 0; jj < 8; ++jj)
        ap[gg * 8 + jj] = sP[lo * 32 + hi * 16 + gg * 8 + jj];

    // V tile for this step must have landed
    if (cs < 15) __builtin_amdgcn_s_wait_tensorcnt(1);
    else         __builtin_amdgcn_s_wait_tensorcnt(0);
    const bf16* vb = (const bf16*)(lds_raw + ((cs & 1) ? LDS_SV1 : LDS_SV0));
#pragma unroll
    for (int dt = 0; dt < 4; ++dt)
      oacc[dt] = wmma_bf16(ap, load_b_lds_kn(vb, dt * 16), oacc[dt]);
    __syncthreads();
  }

  // normalize and store bf16 for the output projection
#pragma unroll
  for (int dt = 0; dt < 4; ++dt)
#pragma unroll
    for (int j = 0; j < 8; ++j) {
      const float o = oacc[dt][j] / l_run[j];
      const int n = n0 + j + 8 * hi;
      const int d = dt * 16 + lo;
      attnb[((long)n * N_GRAPH + g) * E_DIM + h * HD + d] = (bf16)o;
    }
}

// ---------------------------------------------------------------------------
// Output projection: out[m,e] = attn[m,:] @ out_w[e,:] + out_b[e]   (f32 out)
// 16x64 tile per wave.
// ---------------------------------------------------------------------------
__global__ void __launch_bounds__(256) out_gemm_kernel(
    const bf16* __restrict__ A, const bf16* __restrict__ W,
    const float* __restrict__ bias, float* __restrict__ out) {
  const int wid = threadIdx.x >> 5;
  const int n0 = (blockIdx.x * 8 + wid) * 64;   // 0..960
  const int m0 = blockIdx.y * 16;               // 0..4080
  v8f c[4] = {};
  for (int k0 = 0; k0 < E_DIM; k0 += 32) {
    v16bf a = load_a_rowmajor(A, m0, E_DIM, k0);
#pragma unroll
    for (int t = 0; t < 4; ++t)
      c[t] = wmma_bf16(a, load_b_nk(W, n0 + t * 16, E_DIM, k0), c[t]);
  }
  const int lane = threadIdx.x & 31;
  const int hi = lane >> 4;
#pragma unroll
  for (int t = 0; t < 4; ++t) {
    const int n = n0 + t * 16 + (lane & 15);
    const float bv = bias[n];
#pragma unroll
    for (int j = 0; j < 8; ++j) {
      int m = m0 + j + 8 * hi;
      out[(long)m * E_DIM + n] = c[t][j] + bv;
    }
  }
}

// ---------------------------------------------------------------------------
extern "C" void kernel_launch(void* const* d_in, const int* in_sizes, int n_in,
                              void* d_out, int out_size, void* d_ws, size_t ws_size,
                              hipStream_t stream) {
  const float* query     = (const float*)d_in[0];   // [512, 8, 1024]
  const float* attn_bias = (const float*)d_in[1];   // [128, 512, 512]
  const float* gw        = (const float*)d_in[2];   // [128, 512, 512]
  const float* in_w      = (const float*)d_in[3];   // [3072, 1024]
  const float* in_b      = (const float*)d_in[4];   // [3072]
  const float* out_w     = (const float*)d_in[5];   // [1024, 1024]
  const float* out_b     = (const float*)d_in[6];   // [1024]

  bf16* qbf   = (bf16*)d_ws;                          // 4096*1024
  bf16* wbf   = qbf  + (size_t)M_ROWS * E_DIM;        // 3072*1024
  bf16* owbf  = wbf  + (size_t)QKV_LD * E_DIM;        // 1024*1024
  bf16* qkv   = owbf + (size_t)E_DIM * E_DIM;         // 4096*3072
  bf16* attnb = qkv  + (size_t)M_ROWS * QKV_LD;       // 4096*1024

  const int nq4 = M_ROWS * E_DIM / 4;
  const int nw4 = QKV_LD * E_DIM / 4;
  const int no4 = E_DIM * E_DIM / 4;
  cvt_bf16_kernel<<<(nq4 + 255) / 256, 256, 0, stream>>>(
      (const float4*)query, (v4bf*)qbf, nq4);
  cvt_bf16_kernel<<<(nw4 + 255) / 256, 256, 0, stream>>>(
      (const float4*)in_w, (v4bf*)wbf, nw4);
  cvt_bf16_kernel<<<(no4 + 255) / 256, 256, 0, stream>>>(
      (const float4*)out_w, (v4bf*)owbf, no4);

  // QKV GEMM: (3072/64)/8 x 4096/16
  qkv_gemm_kernel<<<dim3(QKV_LD / 512, M_ROWS / 16), 256, 0, stream>>>(
      qbf, wbf, in_b, qkv);

  // Fused attention: one wave per (batch, 16-row tile)
  attention_kernel<<<NBATCH * (N_NODE / 16), 32, 0, stream>>>(
      qkv, attn_bias, gw, attnb);

  // Output GEMM: (1024/64)/8 x 4096/16
  out_gemm_kernel<<<dim3(E_DIM / 512, M_ROWS / 16), 256, 0, stream>>>(
      attnb, owbf, out_b, (float*)d_out);
}
